// ConceptGenerator_91199335563520
// MI455X (gfx1250) — matrix operand order repaired and checked by reading, not compiled, not run on hardware
//
#include <hip/hip_runtime.h>
#include <math.h>

#define SEQ 128
#define BSZ 32
#define SRCLEN 256
#define DDIM 512
#define CDIM 512
#define VDIM 40000
#define ROWS (SEQ*BSZ)      /* 4096 */
#define KC16 16             /* K/32 for K=512 */
#define NT_V (VDIM/16)      /* 2500 */
#define PAD_IDX 0

typedef __bf16 bf16;
typedef __attribute__((ext_vector_type(16))) __bf16 bf16x16;
typedef __attribute__((ext_vector_type(8)))  __bf16 bf16x8;
typedef __attribute__((ext_vector_type(8)))  float  f32x8;

// ---------------------------------------------------------------------------
// A-fragment loader: CDNA5 16x32 bf16 A layout. Lane L (group g=L>>4) holds
// row M=L&15, K = {0..7,16..23} (g=0) or {8..15,24..31} (g=1): two contiguous
// 8-element segments of a row-major K-line -> two 16B loads.
// ---------------------------------------------------------------------------
__device__ __forceinline__ bf16x16 load_a_frag(const bf16* __restrict__ arow, int kc) {
    bf16x8 s0 = *(const bf16x8*)(arow + kc * 32);
    bf16x8 s1 = *(const bf16x8*)(arow + kc * 32 + 16);
    return __builtin_shufflevector(s0, s1, 0,1,2,3,4,5,6,7,8,9,10,11,12,13,14,15);
}

// f32 -> bf16 elementwise (row-major)
__global__ void k_cvt_bf16(const float* __restrict__ x, bf16* __restrict__ y, int n) {
    int i = blockIdx.x * blockDim.x + threadIdx.x;
    if (i < n) y[i] = (bf16)x[i];
}

// f32 KxN row-major  ->  bf16 B fragments, tile-major:
// idx = ((nt*KC + kc)*32 + lane)*16 + e,  lane = (n&15) + 16*((kk>>3)&1), e = (kk&7) + 8*(kk>>4)
__global__ void k_swizzle_b(const float* __restrict__ w, bf16* __restrict__ f,
                            int K, int N) {
    int total = K * N;
    int KC = K >> 5, stride = gridDim.x * blockDim.x;
    for (int i = blockIdx.x * blockDim.x + threadIdx.x; i < total; i += stride) {
        int k = i / N, n = i - k * N;
        int kk = k & 31, kc = k >> 5;
        int g = (kk >> 3) & 1, e = (kk & 7) + ((kk >> 4) << 3);
        int nt = n >> 4, lane = (n & 15) + (g << 4);
        f[(((size_t)nt * KC + kc) * 32 + lane) * 16 + e] = (bf16)w[i];
    }
}

// oc = tanh(outs @ W_t + b_t), bf16 WMMA, writes oc row-major bf16
__global__ void __launch_bounds__(256)
k_gemm_tanh(const bf16* __restrict__ Abf, const bf16* __restrict__ Bfrag,
            const float* __restrict__ bias, bf16* __restrict__ Obf) {
    const int lane = threadIdx.x & 31;
    const int wid  = threadIdx.x >> 5;
    const int m0 = blockIdx.x * 16;
    const int nt = blockIdx.y * 8 + wid;
    const bf16* arow = Abf + (size_t)(m0 + (lane & 15)) * DDIM + ((lane >> 4) << 3);
    const bf16* bptr = Bfrag + (size_t)nt * (KC16 * 512) + lane * 16;
    f32x8 acc = {};
    #pragma unroll
    for (int kc = 0; kc < KC16; ++kc) {
        bf16x16 a = load_a_frag(arow, kc);
        bf16x16 b = *(const bf16x16*)(bptr + kc * 512);
        acc = __builtin_amdgcn_wmma_f32_16x16x32_bf16(false, a, false, b,
                                                      (short)0, acc, false, false);
    }
    const int col = nt * 16 + (lane & 15);
    const float bt = bias[col];
    const int rbase = m0 + ((lane >> 4) << 3);
    #pragma unroll
    for (int j = 0; j < 8; ++j)
        Obf[(size_t)(rbase + j) * CDIM + col] = (bf16)tanhf(acc[j] + bt);
}

// gates = softmax(oc @ W_d + b_d); stored per row as [gen, g2, g1]
__global__ void k_gates(const bf16* __restrict__ ocbf, const float* __restrict__ Wd,
                        const float* __restrict__ bd, float* __restrict__ gates) {
    int r = blockIdx.x * blockDim.x + threadIdx.x;
    if (r >= ROWS) return;
    float l0 = bd[0], l1 = bd[1], l2 = bd[2];
    const bf16* oc = ocbf + (size_t)r * CDIM;
    for (int c = 0; c < CDIM; ++c) {
        float o = (float)oc[c];
        l0 += o * Wd[c * 3 + 0];
        l1 += o * Wd[c * 3 + 1];
        l2 += o * Wd[c * 3 + 2];
    }
    float m = fmaxf(l0, fmaxf(l1, l2));
    float e0 = __expf(l0 - m), e1 = __expf(l1 - m), e2 = __expf(l2 - m);
    float inv = 1.f / (e0 + e1 + e2);
    gates[r * 3 + 0] = e0 * inv;   // gen gate
    gates[r * 3 + 1] = e2 * inv;   // copy gate for p even (gates[...,2])
    gates[r * 3 + 2] = e1 * inv;   // copy gate for p odd  (gates[...,1])
}

// ---------------------------------------------------------------------------
// Flash-softmax over V, 32 rows per block: two resident A-fragment sets share
// each streamed B fragment (1 b128-load pair feeds 2 WMMAs); halves W_g L2
// traffic vs the 16-row version. ~330 VGPRs/wave (gfx1250 allows 1024).
// ---------------------------------------------------------------------------
__global__ void __launch_bounds__(256)
k_softmax_sweep(const bf16* __restrict__ ocbf, const bf16* __restrict__ Wgfrag,
                const float* __restrict__ bg, float* __restrict__ gmax,
                float* __restrict__ gsum) {
    const int lane = threadIdx.x & 31;
    const int wid  = threadIdx.x >> 5;
    const int m0 = blockIdx.x * 32;

    bf16x16 afragA[KC16], afragB[KC16];
    {
        const bf16* arowA = ocbf + (size_t)(m0 + (lane & 15)) * CDIM + ((lane >> 4) << 3);
        const bf16* arowB = arowA + (size_t)16 * CDIM;
        #pragma unroll
        for (int kc = 0; kc < KC16; ++kc) {
            afragA[kc] = load_a_frag(arowA, kc);
            afragB[kc] = load_a_frag(arowB, kc);
        }
    }

    float mA[8], sA[8], mB[8], sB[8];
    #pragma unroll
    for (int j = 0; j < 8; ++j) { mA[j] = -1e30f; sA[j] = 0.f; mB[j] = -1e30f; sB[j] = 0.f; }

    for (int nt = wid; nt < NT_V; nt += 8) {
        const bf16* bptr = Wgfrag + (size_t)nt * (KC16 * 512) + lane * 16;
        if (nt + 8 < NT_V)  // pull next 16KB tile toward L2/L0
            __builtin_prefetch(Wgfrag + (size_t)(nt + 8) * (KC16 * 512) + lane * 512, 0, 1);
        f32x8 accA = {}, accB = {};
        #pragma unroll
        for (int kc = 0; kc < KC16; ++kc) {
            bf16x16 b = *(const bf16x16*)(bptr + kc * 512);
            accA = __builtin_amdgcn_wmma_f32_16x16x32_bf16(false, afragA[kc], false, b,
                                                           (short)0, accA, false, false);
            accB = __builtin_amdgcn_wmma_f32_16x16x32_bf16(false, afragB[kc], false, b,
                                                           (short)0, accB, false, false);
        }
        const float bias = bg[nt * 16 + (lane & 15)];
        #pragma unroll
        for (int j = 0; j < 8; ++j) {
            float zA = accA[j] + bias;
            float nmA = fmaxf(mA[j], zA);
            sA[j] = sA[j] * __expf(mA[j] - nmA) + __expf(zA - nmA);
            mA[j] = nmA;
            float zB = accB[j] + bias;
            float nmB = fmaxf(mB[j], zB);
            sB[j] = sB[j] * __expf(mB[j] - nmB) + __expf(zB - nmB);
            mB[j] = nmB;
        }
    }

    // cross-lane merge within each 16-lane group (columns of the same row)
    #pragma unroll
    for (int off = 1; off < 16; off <<= 1) {
        #pragma unroll
        for (int j = 0; j < 8; ++j) {
            float om, os, nm;
            om = __shfl_xor(mA[j], off, 32); os = __shfl_xor(sA[j], off, 32);
            nm = fmaxf(mA[j], om);
            sA[j] = sA[j] * __expf(mA[j] - nm) + os * __expf(om - nm);
            mA[j] = nm;
            om = __shfl_xor(mB[j], off, 32); os = __shfl_xor(sB[j], off, 32);
            nm = fmaxf(mB[j], om);
            sB[j] = sB[j] * __expf(mB[j] - nm) + os * __expf(om - nm);
            mB[j] = nm;
        }
    }

    __shared__ float red_m[8][32];
    __shared__ float red_s[8][32];
    if ((lane & 15) == 0) {
        int g = lane >> 4;
        #pragma unroll
        for (int j = 0; j < 8; ++j) {
            red_m[wid][g * 8 + j]      = mA[j];
            red_s[wid][g * 8 + j]      = sA[j];
            red_m[wid][16 + g * 8 + j] = mB[j];
            red_s[wid][16 + g * 8 + j] = sB[j];
        }
    }
    __syncthreads();
    if (threadIdx.x < 32) {
        float fm = -1e30f, fs = 0.f;
        for (int w = 0; w < 8; ++w) {
            float om = red_m[w][threadIdx.x], os = red_s[w][threadIdx.x];
            float nm = fmaxf(fm, om);
            fs = fs * __expf(fm - nm) + os * __expf(om - nm);
            fm = nm;
        }
        gmax[m0 + threadIdx.x] = fm;
        gsum[m0 + threadIdx.x] = fs;
    }
}

// Per-(s,b) loss: target logit (same bf16 operands via fragment indexing),
// gen prob + copy-scatter contribution at the target, -log, PAD mask.
__global__ void k_loss_row(const bf16* __restrict__ ocbf, const bf16* __restrict__ Wgfrag,
                           const float* __restrict__ bg, const float* __restrict__ gmax,
                           const float* __restrict__ gsum, const float* __restrict__ gates,
                           const int* __restrict__ copy_seq, const int* __restrict__ target,
                           const float* __restrict__ aw, float* __restrict__ lossrow) {
    int r = blockIdx.x * blockDim.x + threadIdx.x;
    if (r >= ROWS) return;
    const int b = r & 31;          // r = s*BSZ + b
    const int t = target[r];

    float z = bg[t];
    const bf16* oc  = ocbf + (size_t)r * CDIM;
    const int nt = t >> 4, ln = t & 15;
    const bf16* wcol = Wgfrag + (size_t)nt * (KC16 * 512);
    for (int c = 0; c < CDIM; ++c) {
        int kk = c & 31, kc = c >> 5;
        int lane = ln + (((kk >> 3) & 1) << 4), e = (kk & 7) + ((kk >> 4) << 3);
        z += (float)oc[c] * (float)wcol[((size_t)kc * 32 + lane) * 16 + e];
    }
    float pgen = gates[r * 3 + 0] * __expf(z - gmax[r]) / gsum[r];

    float g_even = gates[r * 3 + 1], g_odd = gates[r * 3 + 2];
    const float* awr = aw + (size_t)r * SRCLEN;
    float cadd = 0.f;
    for (int p = 0; p < 2 * SRCLEN; ++p) {
        if (copy_seq[p * BSZ + b] == t)
            cadd += ((p & 1) == 0 ? g_even : g_odd) * awr[p >> 1];
    }
    float prob = pgen + cadd;
    lossrow[r] = (t == PAD_IDX) ? 0.f : -logf(prob + 1e-12f);
}

__global__ void k_loss_reduce(const float* __restrict__ lossrow, float* __restrict__ out) {
    int b = threadIdx.x;
    if (b < BSZ) {
        float s = 0.f;
        for (int sq = 0; sq < SEQ; ++sq) s += lossrow[sq * BSZ + b];
        out[b] = s;
    }
}

extern "C" void kernel_launch(void* const* d_in, const int* in_sizes, int n_in,
                              void* d_out, int out_size, void* d_ws, size_t ws_size,
                              hipStream_t stream) {
    const float* outs   = (const float*)d_in[0];
    const float* aw     = (const float*)d_in[1];
    const int*   cseq   = (const int*)  d_in[2];
    const int*   target = (const int*)  d_in[3];
    const float* W_t    = (const float*)d_in[4];
    const float* b_t    = (const float*)d_in[5];
    const float* W_g    = (const float*)d_in[6];
    const float* b_g    = (const float*)d_in[7];
    const float* W_d    = (const float*)d_in[8];
    const float* b_d    = (const float*)d_in[9];
    float* out = (float*)d_out;

    char* ws = (char*)d_ws;
    bf16*  wgfrag  = (bf16*) (ws + 0);                 // 512*40000*2 = 40,960,000
    bf16*  outsbf  = (bf16*) (ws + 40960000);          // 4,194,304
    bf16*  wtfrag  = (bf16*) (ws + 45154304);          // 524,288
    bf16*  ocbf    = (bf16*) (ws + 45678592);          // 4,194,304
    float* gates   = (float*)(ws + 49872896);          // 49,152
    float* gmax    = (float*)(ws + 49922048);          // 16,384
    float* gsum    = (float*)(ws + 49938432);          // 16,384
    float* lossrow = (float*)(ws + 49954816);          // 16,384

    // 1) precision conversion + B-fragment swizzles
    k_cvt_bf16<<<(ROWS * DDIM + 255) / 256, 256, 0, stream>>>(outs, outsbf, ROWS * DDIM);
    k_swizzle_b<<<(DDIM * CDIM + 255) / 256, 256, 0, stream>>>(W_t, wtfrag, DDIM, CDIM);
    k_swizzle_b<<<(CDIM * VDIM + 255) / 256, 256, 0, stream>>>(W_g, wgfrag, CDIM, VDIM);

    // 2) oc = tanh(outs @ W_t + b_t)   [WMMA bf16]
    k_gemm_tanh<<<dim3(ROWS / 16, CDIM / 128), 256, 0, stream>>>(outsbf, wtfrag, b_t, ocbf);

    // 3) 3-way gates
    k_gates<<<(ROWS + 255) / 256, 256, 0, stream>>>(ocbf, W_d, b_d, gates);

    // 4) flash-softmax normalizers over V=40000   [WMMA bf16, 32 rows/block]
    k_softmax_sweep<<<ROWS / 32, 256, 0, stream>>>(ocbf, wgfrag, b_g, gmax, gsum);

    // 5) per-row loss (target logit + copy scatter at target)
    k_loss_row<<<(ROWS + 255) / 256, 256, 0, stream>>>(ocbf, wgfrag, b_g, gmax, gsum,
                                                       gates, cseq, target, aw, lossrow);

    // 6) reduce over SEQ (fixed order -> deterministic), then pass-through outs
    k_loss_reduce<<<1, 32, 0, stream>>>(lossrow, out);
    hipMemcpyAsync(out + BSZ, outs, (size_t)ROWS * DDIM * sizeof(float),
                   hipMemcpyDeviceToDevice, stream);
}
